// SelfAttentionLayer_74096775790973
// MI455X (gfx1250) — matrix operand order repaired and checked by reading, not compiled, not run on hardware
//
#include <hip/hip_runtime.h>
#include <hip/hip_bf16.h>

typedef __bf16 bf16_t;
typedef __attribute__((ext_vector_type(16))) __bf16 v16bf;
typedef __attribute__((ext_vector_type(8)))  float  v8f;
typedef __attribute__((ext_vector_type(4)))  unsigned int u32x4;
typedef __attribute__((ext_vector_type(8)))  int          i32x8;
typedef __attribute__((ext_vector_type(4)))  int          i32x4;

#define B_   4
#define S_   2048
#define E_   1024
#define H_   16
#define D_   64
#define M_   (B_ * S_)          // 8192 rows
#define EPS_ 1e-6f

// ---------------------------------------------------------------------------
// WMMA helpers (CDNA5 16x16x32 bf16, f32 accumulate). VGPR layouts per ISA:
//  A (16x32): lane L holds row M=L%16; K = (L<16?0:8)+i for i<8, +16 for i>=8
//  B (32x16): lane L holds col N=L%16; K = (L<16?0:8)+i for i<8, +16 for i>=8
//  C/D (16x16 f32): elem v -> row M = (L<16? v : v+8), col N = L%16
// ---------------------------------------------------------------------------
__device__ __forceinline__ v8f wmma_bf16(v16bf a, v16bf b, v8f c) {
  return __builtin_amdgcn_wmma_f32_16x16x32_bf16(false, a, false, b,
                                                 (short)0, c, false, false);
}

// A fragment from a row-major [16 x >=kbase+32] bf16 tile (16B-aligned rows).
__device__ __forceinline__ v16bf load_a_frag(const bf16_t* base, int row_stride,
                                             int lane, int kbase) {
  int m  = lane & 15;
  int k0 = kbase + ((lane < 16) ? 0 : 8);
  union { v16bf v; uint4 u[2]; } r;
  r.u[0] = *reinterpret_cast<const uint4*>(base + m * row_stride + k0);
  r.u[1] = *reinterpret_cast<const uint4*>(base + m * row_stride + k0 + 16);
  return r.v;
}

// B fragment where B^T is stored row-major: element i -> base[n*stride + k0+i]
// for i<8, base[n*stride + k0+16+(i-8)] for i>=8 (memory holds B(k,n) at [n][k]).
__device__ __forceinline__ v16bf load_b_frag_contig(const bf16_t* base,
                                                    int row_stride, int n,
                                                    int lane, int kbase) {
  int k0 = kbase + ((lane < 16) ? 0 : 8);
  union { v16bf v; uint4 u[2]; } r;
  const uint4* p = reinterpret_cast<const uint4*>(base + n * row_stride + k0);
  r.u[0] = p[0];
  r.u[1] = p[2];
  return r.v;
}

// ---------------------------------------------------------------------------
// TDM: DMA a 64x64 bf16 tile (row stride 64 elems in global) into LDS with
// hardware padding to a [64][72] bf16 layout (128B rows + 16B pad).
// D# bitfields per CDNA5 ISA.
// Issued by one wave; completion via TENSORcnt.
// ---------------------------------------------------------------------------
__device__ __forceinline__ void tdm_load_tile_64x64_pad(unsigned lds_byte_addr,
                                                        const bf16_t* gptr) {
  unsigned long long ga = (unsigned long long)(size_t)gptr;
  u32x4 g0;
  g0[0] = 1u;                                   // count=1 (valid), user mode
  g0[1] = lds_byte_addr;                        // lds_addr
  g0[2] = (unsigned)(ga & 0xFFFFFFFFu);         // global_addr[31:0]
  g0[3] = (unsigned)((ga >> 32) & 0x1FFFFFFu)   // global_addr[56:32]
          | (2u << 30);                         // type = 2 ("image")
  i32x8 g1;
  g1[0] = (int)((1u << 16)      // data_size = 1 -> 2 bytes/elem
              | (1u << 20)      // pad_enable
              | (4u << 22)      // pad_interval code 4 -> every 32 DWORDs (128B)
              | (3u << 25));    // pad_amount  code 3 -> 4 DWORDs (16B)
  g1[1] = (int)(64u << 16);     // tensor_dim0[15:0]=64 in bits[63:48]
  g1[2] = (int)(64u << 16);     // tensor_dim0 hi=0 | tensor_dim1[15:0]=64
  g1[3] = (int)(64u << 16);     // tensor_dim1 hi=0 | tile_dim0=64
  g1[4] = 64;                   // tile_dim1=64, tile_dim2=0
  g1[5] = 64;                   // tensor_dim0_stride = 64 (elements)
  g1[6] = 0;                    // stride0 hi / tensor_dim1_stride lo
  g1[7] = 0;                    // tensor_dim1_stride hi
  i32x4 gz = {0, 0, 0, 0};
  asm volatile("tensor_load_to_lds %0, %1, %2, %3"
               :
               : "s"(g0), "s"(g1), "s"(gz), "s"(gz)
               : "memory");
}

// ---------------------------------------------------------------------------
// f32 -> bf16 conversion
// ---------------------------------------------------------------------------
__global__ __launch_bounds__(256) void f32_to_bf16_kernel(
    const float* __restrict__ src, bf16_t* __restrict__ dst, int n) {
  int i = blockIdx.x * 256 + threadIdx.x;
  if (i < n) dst[i] = (bf16_t)src[i];
}

// ---------------------------------------------------------------------------
// QKV projection GEMM: Y[8192,1024] = X @ W + bias, output bf16 [B,H,S,D].
// Block tile 64(M) x 128(N), K step 32; W tile staged TRANSPOSED in LDS so
// both A and B fragments are contiguous ds_load_b128 pairs (no scalar packs).
// ---------------------------------------------------------------------------
__global__ __launch_bounds__(256) void gemm_qkv_kernel(
    const bf16_t* __restrict__ X, const bf16_t* __restrict__ W,
    const float* __restrict__ bias, bf16_t* __restrict__ Out) {
  __shared__ bf16_t As[64][32];
  __shared__ bf16_t Bt[128][40];   // W^T tile: [n][k], 80B row (16B aligned)

  const int tid  = threadIdx.x;
  const int lane = tid & 31;
  const int wave = tid >> 5;
  const int wm   = wave & 1;   // 2 wave rows (32 rows each)
  const int wn   = wave >> 1;  // 4 wave cols (32 cols each)
  const int m0   = blockIdx.y * 64;
  const int n0   = blockIdx.x * 128;

  const int a_row = tid >> 2;        // 0..63
  const int a_col = (tid & 3) * 8;   // 0..24

  v8f acc[2][2] = {};

  for (int k0 = 0; k0 < E_; k0 += 32) {
    __syncthreads();
    *reinterpret_cast<uint4*>(&As[a_row][a_col]) =
        *reinterpret_cast<const uint4*>(X + (size_t)(m0 + a_row) * E_ + k0 + a_col);
#pragma unroll
    for (int j = 0; j < 2; ++j) {
      int idx = tid + j * 256;       // 0..511
      int r = idx >> 4;              // k: 0..31
      int c = (idx & 15) * 8;        // n: 0..120
      union { uint4 u; bf16_t e[8]; } w;
      w.u = *reinterpret_cast<const uint4*>(W + (size_t)(k0 + r) * E_ + n0 + c);
#pragma unroll
      for (int i = 0; i < 8; ++i) Bt[c + i][r] = w.e[i];
    }
    if (k0 + 32 < E_) {
      __builtin_prefetch(X + (size_t)(m0 + a_row) * E_ + k0 + 32 + a_col, 0, 1);
      __builtin_prefetch(W + (size_t)(k0 + 32 + (tid >> 4)) * E_ + n0 + (tid & 15) * 8, 0, 1);
    }
    __syncthreads();

    v16bf af[2], bfr[2];
#pragma unroll
    for (int mi = 0; mi < 2; ++mi)
      af[mi] = load_a_frag(&As[wm * 32 + mi * 16][0], 32, lane, 0);
#pragma unroll
    for (int ni = 0; ni < 2; ++ni)
      bfr[ni] = load_b_frag_contig(&Bt[0][0], 40,
                                   wn * 32 + ni * 16 + (lane & 15), lane, 0);
#pragma unroll
    for (int mi = 0; mi < 2; ++mi)
#pragma unroll
      for (int ni = 0; ni < 2; ++ni)
        acc[mi][ni] = wmma_bf16(af[mi], bfr[ni], acc[mi][ni]);
  }

  // epilogue: +bias, store bf16 head-split [B,H,S,D]
#pragma unroll
  for (int mi = 0; mi < 2; ++mi) {
#pragma unroll
    for (int ni = 0; ni < 2; ++ni) {
      int n = n0 + wn * 32 + ni * 16 + (lane & 15);
      float bn = bias[n];
      int h = n >> 6, d = n & 63;
#pragma unroll
      for (int v = 0; v < 8; ++v) {
        int m  = m0 + wm * 32 + mi * 16 + ((lane < 16) ? v : v + 8);
        int bb = m >> 11;       // / S_
        int s  = m & (S_ - 1);
        Out[(((size_t)bb * H_ + h) * S_ + s) * D_ + d] =
            (bf16_t)(acc[mi][ni][v] + bn);
      }
    }
  }
}

// ---------------------------------------------------------------------------
// Output projection GEMM: Y = Ctx @ Wo + bo + x (residual), f32 out [M,E].
// ---------------------------------------------------------------------------
__global__ __launch_bounds__(256) void gemm_out_kernel(
    const bf16_t* __restrict__ Ctx, const bf16_t* __restrict__ W,
    const float* __restrict__ bias, const float* __restrict__ Xres,
    float* __restrict__ Y) {
  __shared__ bf16_t As[64][32];
  __shared__ bf16_t Bt[128][40];

  const int tid  = threadIdx.x;
  const int lane = tid & 31;
  const int wave = tid >> 5;
  const int wm   = wave & 1;
  const int wn   = wave >> 1;
  const int m0   = blockIdx.y * 64;
  const int n0   = blockIdx.x * 128;

  const int a_row = tid >> 2;
  const int a_col = (tid & 3) * 8;

  v8f acc[2][2] = {};

  for (int k0 = 0; k0 < E_; k0 += 32) {
    __syncthreads();
    *reinterpret_cast<uint4*>(&As[a_row][a_col]) =
        *reinterpret_cast<const uint4*>(Ctx + (size_t)(m0 + a_row) * E_ + k0 + a_col);
#pragma unroll
    for (int j = 0; j < 2; ++j) {
      int idx = tid + j * 256;
      int r = idx >> 4;
      int c = (idx & 15) * 8;
      union { uint4 u; bf16_t e[8]; } w;
      w.u = *reinterpret_cast<const uint4*>(W + (size_t)(k0 + r) * E_ + n0 + c);
#pragma unroll
      for (int i = 0; i < 8; ++i) Bt[c + i][r] = w.e[i];
    }
    if (k0 + 32 < E_) {
      __builtin_prefetch(Ctx + (size_t)(m0 + a_row) * E_ + k0 + 32 + a_col, 0, 1);
      __builtin_prefetch(W + (size_t)(k0 + 32 + (tid >> 4)) * E_ + n0 + (tid & 15) * 8, 0, 1);
    }
    __syncthreads();

    v16bf af[2], bfr[2];
#pragma unroll
    for (int mi = 0; mi < 2; ++mi)
      af[mi] = load_a_frag(&As[wm * 32 + mi * 16][0], 32, lane, 0);
#pragma unroll
    for (int ni = 0; ni < 2; ++ni)
      bfr[ni] = load_b_frag_contig(&Bt[0][0], 40,
                                   wn * 32 + ni * 16 + (lane & 15), lane, 0);
#pragma unroll
    for (int mi = 0; mi < 2; ++mi)
#pragma unroll
      for (int ni = 0; ni < 2; ++ni)
        acc[mi][ni] = wmma_bf16(af[mi], bfr[ni], acc[mi][ni]);
  }

#pragma unroll
  for (int mi = 0; mi < 2; ++mi) {
#pragma unroll
    for (int ni = 0; ni < 2; ++ni) {
      int n = n0 + wn * 32 + ni * 16 + (lane & 15);
      float bn = bias[n];
#pragma unroll
      for (int v = 0; v < 8; ++v) {
        int m = m0 + wm * 32 + mi * 16 + ((lane < 16) ? v : v + 8);
        size_t idx = (size_t)m * E_ + n;
        Y[idx] = acc[mi][ni][v] + bn + Xres[idx];
      }
    }
  }
}

// ---------------------------------------------------------------------------
// Flash attention per (b,h): 64-query-row tile per block, key tiles of 64.
// K tile staged by the Tensor Data Mover (hardware-padded LDS layout);
// scores = Q K^T / 32 + mask*(-1e9); online softmax; O += P V.
// ---------------------------------------------------------------------------
__global__ __launch_bounds__(256) void attention_kernel(
    const bf16_t* __restrict__ Q, const bf16_t* __restrict__ K,
    const bf16_t* __restrict__ V, const float* __restrict__ mask,
    bf16_t* __restrict__ Ctx) {
  __shared__ bf16_t Qs[64][72];   // [row][d]
  __shared__ bf16_t Ks[64][72];   // [key][d]   (B^T layout for QK^T; TDM dest)
  __shared__ bf16_t Vt[64][72];   // [d][key]   (B^T layout for P V)
  __shared__ bf16_t Ps[64][72];   // [row][key] probabilities (bf16)
  __shared__ float  Sf[64][68];   // [row][key] raw scores (f32)
  __shared__ float  row_m[64], row_l[64], row_scale[64];

  const int tid  = threadIdx.x;
  const int lane = tid & 31;
  const int wave = tid >> 5;
  const int wm   = wave >> 1;  // row group (16 rows)
  const int wn   = wave & 1;   // col group (32 cols)
  const int q0   = blockIdx.x * 64;
  const int bh   = blockIdx.y;      // b*H + h
  const int bb   = bh >> 4;
  const int h    = bh & 15;

  const bf16_t* Qg = Q + (size_t)bh * S_ * D_;
  const bf16_t* Kg = K + (size_t)bh * S_ * D_;
  const bf16_t* Vg = V + (size_t)bh * S_ * D_;
  const float*  mrow = mask + (size_t)bb * S_;
  const unsigned ks_lds = (unsigned)(size_t)(&Ks[0][0]);

  // stage the query tile (64 x 64 bf16)
#pragma unroll
  for (int j = 0; j < 2; ++j) {
    int idx = tid + j * 256;
    int r = idx >> 3, c = (idx & 7) * 8;
    *reinterpret_cast<uint4*>(&Qs[r][c]) =
        *reinterpret_cast<const uint4*>(Qg + (size_t)(q0 + r) * D_ + c);
  }
  if (tid < 64) { row_m[tid] = -INFINITY; row_l[tid] = 0.f; }

  v8f o[2] = {};  // output accumulator: rows wm*16.., d-cols wn*32 + ni*16

  for (int kt = 0; kt < S_; kt += 64) {
    __syncthreads();  // previous iter done with Ks/Vt/Ps; Q/stats init visible

    // K tile via Tensor Data Mover (one wave issues; TENSORcnt tracks it)
    if (wave == 0) {
      tdm_load_tile_64x64_pad(ks_lds, Kg + (size_t)kt * D_);
      __builtin_amdgcn_s_wait_tensorcnt((short)0);
    }
    // V tile transposed [d][key] (vector path)
#pragma unroll
    for (int j = 0; j < 16; ++j) {
      int idx = tid + j * 256;
      int key = idx >> 6, d = idx & 63;
      Vt[d][key] = Vg[(size_t)(kt + key) * D_ + d];
    }
    __syncthreads();

    // ---- scores: S = Q K^T (contraction over d, 2 x K=32 steps) ----
    v8f sacc[2] = {};
#pragma unroll
    for (int kk = 0; kk < 2; ++kk) {
      v16bf aq = load_a_frag(&Qs[wm * 16][0], 72, lane, kk * 32);
#pragma unroll
      for (int ni = 0; ni < 2; ++ni) {
        v16bf bk = load_b_frag_contig(&Ks[0][0], 72,
                                      wn * 32 + ni * 16 + (lane & 15), lane,
                                      kk * 32);
        sacc[ni] = wmma_bf16(aq, bk, sacc[ni]);
      }
    }
#pragma unroll
    for (int ni = 0; ni < 2; ++ni) {
      int col  = wn * 32 + ni * 16 + (lane & 15);
      float mk = mrow[kt + col] * (-1e9f);
#pragma unroll
      for (int v = 0; v < 8; ++v) {
        int row = wm * 16 + ((lane < 16) ? v : v + 8);
        Sf[row][col] = sacc[ni][v] * 0.03125f + mk;  // 1/sqrt(E)=1/32
      }
    }
    __syncthreads();

    // ---- online softmax stats: 4 threads per row ----
    {
      int row = tid >> 2;
      int c0  = (tid & 3) * 16;
      float mx = -INFINITY;
#pragma unroll
      for (int i = 0; i < 16; ++i) mx = fmaxf(mx, Sf[row][c0 + i]);
      mx = fmaxf(mx, __shfl_xor(mx, 1, 32));
      mx = fmaxf(mx, __shfl_xor(mx, 2, 32));
      float m_new = fmaxf(row_m[row], mx);
      float sum = 0.f;
#pragma unroll
      for (int i = 0; i < 16; ++i) {
        float e = __expf(Sf[row][c0 + i] - m_new);
        Ps[row][c0 + i] = (bf16_t)e;
        sum += e;
      }
      sum += __shfl_xor(sum, 1, 32);
      sum += __shfl_xor(sum, 2, 32);
      if ((tid & 3) == 0) {
        float sc       = __expf(row_m[row] - m_new);  // 0 on first tile
        row_scale[row] = sc;
        row_l[row]     = row_l[row] * sc + sum;
        row_m[row]     = m_new;
      }
    }
    __syncthreads();

    // ---- rescale running O, then O += P V ----
#pragma unroll
    for (int ni = 0; ni < 2; ++ni)
#pragma unroll
      for (int v = 0; v < 8; ++v) {
        int row = wm * 16 + ((lane < 16) ? v : v + 8);
        o[ni][v] *= row_scale[row];
      }
#pragma unroll
    for (int kk = 0; kk < 2; ++kk) {
      v16bf ap = load_a_frag(&Ps[wm * 16][0], 72, lane, kk * 32);
#pragma unroll
      for (int ni = 0; ni < 2; ++ni) {
        v16bf bv = load_b_frag_contig(&Vt[0][0], 72,
                                      wn * 32 + ni * 16 + (lane & 15), lane,
                                      kk * 32);
        o[ni] = wmma_bf16(ap, bv, o[ni]);
      }
    }
  }

  // ---- normalize and store context, merged-head layout [B,S,E] bf16 ----
#pragma unroll
  for (int ni = 0; ni < 2; ++ni) {
    int d = wn * 32 + ni * 16 + (lane & 15);
#pragma unroll
    for (int v = 0; v < 8; ++v) {
      int row = wm * 16 + ((lane < 16) ? v : v + 8);
      float val = o[ni][v] / row_l[row];
      int s = q0 + row;
      Ctx[((size_t)bb * S_ + s) * E_ + h * D_ + d] = (bf16_t)val;
    }
  }
}

// ---------------------------------------------------------------------------
// LayerNorm per row of 1024 (block = 256 threads, 4 elements each).
// ---------------------------------------------------------------------------
__global__ __launch_bounds__(256) void layernorm_kernel(
    const float* __restrict__ Y, const float* __restrict__ gamma,
    const float* __restrict__ beta, float* __restrict__ out) {
  __shared__ float rs[256], rss[256];
  const int row = blockIdx.x;
  const float* y = Y + (size_t)row * E_;
  float v[4], s = 0.f, ss = 0.f;
#pragma unroll
  for (int j = 0; j < 4; ++j) {
    v[j] = y[threadIdx.x + j * 256];
    s += v[j];
    ss += v[j] * v[j];
  }
  rs[threadIdx.x]  = s;
  rss[threadIdx.x] = ss;
  __syncthreads();
  for (int off = 128; off > 0; off >>= 1) {
    if (threadIdx.x < off) {
      rs[threadIdx.x]  += rs[threadIdx.x + off];
      rss[threadIdx.x] += rss[threadIdx.x + off];
    }
    __syncthreads();
  }
  float mu   = rs[0] * (1.f / E_);
  float var  = rss[0] * (1.f / E_) - mu * mu;
  float rstd = rsqrtf(var + EPS_);
  float* o = out + (size_t)row * E_;
#pragma unroll
  for (int j = 0; j < 4; ++j) {
    int e = threadIdx.x + j * 256;
    o[e] = gamma[e] * (v[j] - mu) * rstd + beta[e];
  }
}

// ---------------------------------------------------------------------------
extern "C" void kernel_launch(void* const* d_in, const int* in_sizes, int n_in,
                              void* d_out, int out_size, void* d_ws,
                              size_t ws_size, hipStream_t stream) {
  const float* x    = (const float*)d_in[0];
  const float* mask = (const float*)d_in[1];
  const float* Wq   = (const float*)d_in[2];
  const float* bq   = (const float*)d_in[3];
  const float* Wk   = (const float*)d_in[4];
  const float* bk   = (const float*)d_in[5];
  const float* Wv   = (const float*)d_in[6];
  const float* bv   = (const float*)d_in[7];
  const float* Wo   = (const float*)d_in[8];
  const float* bo   = (const float*)d_in[9];
  const float* gamma = (const float*)d_in[10];
  const float* beta  = (const float*)d_in[11];
  float* out = (float*)d_out;

  char* ws = (char*)d_ws;
  size_t off = 0;
  bf16_t* xh  = (bf16_t*)(ws + off); off += (size_t)M_ * E_ * 2;   // 16 MB
  bf16_t* Wqh = (bf16_t*)(ws + off); off += (size_t)E_ * E_ * 2;   //  2 MB
  bf16_t* Wkh = (bf16_t*)(ws + off); off += (size_t)E_ * E_ * 2;
  bf16_t* Wvh = (bf16_t*)(ws + off); off += (size_t)E_ * E_ * 2;
  bf16_t* Woh = (bf16_t*)(ws + off); off += (size_t)E_ * E_ * 2;
  bf16_t* Qh  = (bf16_t*)(ws + off); off += (size_t)M_ * E_ * 2;   // 16 MB
  bf16_t* Kh  = (bf16_t*)(ws + off); off += (size_t)M_ * E_ * 2;
  bf16_t* Vh  = (bf16_t*)(ws + off); off += (size_t)M_ * E_ * 2;
  bf16_t* Ctxh = (bf16_t*)(ws + off); off += (size_t)M_ * E_ * 2;  // 16 MB
  // Pre-LN f32 buffer aliases the (dead by then) QKV region: 32 MB <= 48 MB.
  float* Ytmp = (float*)Qh;

  dim3 blk(256);

  const int nX = M_ * E_;
  const int nW = E_ * E_;
  f32_to_bf16_kernel<<<(nX + 255) / 256, blk, 0, stream>>>(x, xh, nX);
  f32_to_bf16_kernel<<<(nW + 255) / 256, blk, 0, stream>>>(Wq, Wqh, nW);
  f32_to_bf16_kernel<<<(nW + 255) / 256, blk, 0, stream>>>(Wk, Wkh, nW);
  f32_to_bf16_kernel<<<(nW + 255) / 256, blk, 0, stream>>>(Wv, Wvh, nW);
  f32_to_bf16_kernel<<<(nW + 255) / 256, blk, 0, stream>>>(Wo, Woh, nW);

  dim3 ggrid(E_ / 128, M_ / 64);  // (8, 128)
  gemm_qkv_kernel<<<ggrid, blk, 0, stream>>>(xh, Wqh, bq, Qh);
  gemm_qkv_kernel<<<ggrid, blk, 0, stream>>>(xh, Wkh, bk, Kh);
  gemm_qkv_kernel<<<ggrid, blk, 0, stream>>>(xh, Wvh, bv, Vh);

  dim3 agrid(S_ / 64, B_ * H_);   // (32, 64)
  attention_kernel<<<agrid, blk, 0, stream>>>(Qh, Kh, Vh, mask, Ctxh);

  gemm_out_kernel<<<ggrid, blk, 0, stream>>>(Ctxh, Woh, bo, x, Ytmp);

  layernorm_kernel<<<dim3(M_), blk, 0, stream>>>(Ytmp, gamma, beta, out);
}
